// ConnectivityGraphGenerator_8924942041826
// MI455X (gfx1250) — compile-verified, matrix operand order checked by reading, not run on hardware
//
#include <hip/hip_runtime.h>
#include <hip/hip_bf16.h>
#include <math.h>

#define B_   128
#define N_   64
#define T_   256
#define IN_  320
#define H_   256
#define OUT_ 128
#define E_   2016
#define BN_  (B_*N_)       // 8192
#define BE_  (B_*E_)       // 258048
#define P4_  512           // combined head width: [mean_top|mean_bot|var_top|var_bot]

typedef __attribute__((ext_vector_type(16))) __bf16 v16bf;
typedef __attribute__((ext_vector_type(8)))  float  v8f;

struct U4 { unsigned int x, y, z, w; };
union ABf { U4 u[2]; v16bf v; };

__device__ inline unsigned short f2bf(float f) {
  unsigned int u = __float_as_uint(f);
  unsigned int r = (u + 0x7FFFu + ((u >> 16) & 1u)) >> 16;   // RNE
  return (unsigned short)r;
}
__device__ inline float bf2f(unsigned short s) {
  return __uint_as_float(((unsigned int)s) << 16);
}
__device__ inline float softplusf(float x) {
  return (x > 20.0f) ? x : log1pf(__expf(x));
}
__device__ inline float wave_max(float m) {
  #pragma unroll
  for (int off = 16; off > 0; off >>= 1) m = fmaxf(m, __shfl_xor(m, off, 32));
  return m;
}
__device__ inline float wave_sum(float s) {
  #pragma unroll
  for (int off = 16; off > 0; off >>= 1) s += __shfl_xor(s, off, 32);
  return s;
}

// ---------------------------------------------------------------------------
// 1) Prefix mean over nodes: agg[b,j,:] = mean(x[b,0..j-1,:]), bf16 out.
//    x = concat(topology[b,j,0:64], temporal[b,j,0:256])  -> 320 features.
// ---------------------------------------------------------------------------
__global__ void prefix_mean_kernel(const float* __restrict__ topo,
                                   const float* __restrict__ temp,
                                   unsigned short* __restrict__ agg) {
  int b = blockIdx.x;
  int f = threadIdx.x;           // 0..319, one feature per thread
  float sum = 0.0f;
  for (int j = 0; j < N_; ++j) {
    float o = (j > 0) ? sum / (float)j : 0.0f;
    agg[(b * N_ + j) * IN_ + f] = f2bf(o);
    float xv = (f < N_) ? topo[(b * N_ + j) * N_ + f]
                        : temp[(b * N_ + j) * T_ + (f - N_)];
    sum += xv;
  }
}

// ---------------------------------------------------------------------------
// 2) Pack weights into the WMMA B-matrix register layout (bf16).
//    Tile = [kt][nt][lane 0..31][16 bf16]; lane l holds column n = nt*16+(l&15),
//    K values {kg*8..kg*8+7, 16+kg*8..16+kg*8+7} with kg = l>>4.
// ---------------------------------------------------------------------------
__global__ void pack_wgnn_kernel(const float* __restrict__ Wg,  // [320,256]
                                 unsigned short* __restrict__ Wp) {
  int tid = blockIdx.x * blockDim.x + threadIdx.x;
  if (tid >= 10 * 16 * 512) return;
  int idx = tid & 15, l = (tid >> 4) & 31, tile = tid >> 9;
  int nt = tile & 15, kt = tile >> 4;
  int n  = nt * 16 + (l & 15);
  int kg = l >> 4;
  int kk = (idx < 8) ? (kg * 8 + idx) : (16 + kg * 8 + (idx - 8));
  int k  = kt * 32 + kk;
  Wp[tid] = f2bf(Wg[k * H_ + n]);
}

__global__ void pack_w4_kernel(const float* __restrict__ Wm,   // [512,128]
                               const float* __restrict__ Wv,   // [512,128]
                               unsigned short* __restrict__ Wp) {
  int tid = blockIdx.x * blockDim.x + threadIdx.x;
  if (tid >= 8 * 32 * 512) return;
  int idx = tid & 15, l = (tid >> 4) & 31, tile = tid >> 9;
  int nt = tile & 31, kt = tile >> 5;
  int c  = nt * 16 + (l & 15);
  int kg = l >> 4;
  int kk = (idx < 8) ? (kg * 8 + idx) : (16 + kg * 8 + (idx - 8));
  int k  = kt * 32 + kk;
  float v;
  if      (c < 128) v = Wm[k * OUT_ + c];               // mean_top
  else if (c < 256) v = Wm[(H_ + k) * OUT_ + (c - 128)];// mean_bot
  else if (c < 384) v = Wv[k * OUT_ + (c - 256)];       // var_top
  else              v = Wv[(H_ + k) * OUT_ + (c - 384)];// var_bot
  Wp[tid] = f2bf(v);
}

// ---------------------------------------------------------------------------
// 3) GEMM1: h = relu(agg @ W_gnn + b_gnn)  [8192,320]x[320,256] -> bf16.
//    One wave per 16x16 output tile; v_wmma_f32_16x16x32_bf16.
// ---------------------------------------------------------------------------
__global__ void gemm1_kernel(const unsigned short* __restrict__ A,   // [8192,320] bf16
                             const unsigned short* __restrict__ Bp,  // packed [10][16][512]
                             const float* __restrict__ bias,         // [256]
                             unsigned short* __restrict__ Hout) {    // [8192,256] bf16
  int wave = (blockIdx.x * blockDim.x + threadIdx.x) >> 5;
  int lane = threadIdx.x & 31;
  int mt = wave >> 4;            // 512 M-tiles
  int nt = wave & 15;            // 16 N-tiles
  int m  = lane & 15;
  int kg = lane >> 4;
  v8f acc = {};
  const unsigned short* arow = A + (mt * 16 + m) * IN_;
  #pragma unroll
  for (int kt = 0; kt < 10; ++kt) {
    ABf a, bmat;
    a.u[0] = *(const U4*)(arow + kt * 32 + kg * 8);
    a.u[1] = *(const U4*)(arow + kt * 32 + 16 + kg * 8);
    const unsigned short* bt = Bp + (kt * 16 + nt) * 512 + lane * 16;
    bmat.u[0] = ((const U4*)bt)[0];
    bmat.u[1] = ((const U4*)bt)[1];
    acc = __builtin_amdgcn_wmma_f32_16x16x32_bf16(false, a.v, false, bmat.v,
                                                  (short)0, acc, false, false);
  }
  int n = nt * 16 + (lane & 15);
  float bv = bias[n];
  int mbase = mt * 16 + 8 * (lane >> 4);
  #pragma unroll
  for (int r = 0; r < 8; ++r) {
    float v = acc[r] + bv;
    v = v > 0.0f ? v : 0.0f;
    Hout[(mbase + r) * H_ + n] = f2bf(v);
  }
}

// ---------------------------------------------------------------------------
// 4) GEMM2: P = h @ W4   [8192,256]x[256,512] -> f32 (biases applied later).
// ---------------------------------------------------------------------------
__global__ void gemm2_kernel(const unsigned short* __restrict__ A,   // [8192,256] bf16
                             const unsigned short* __restrict__ Bp,  // packed [8][32][512]
                             float* __restrict__ P) {                // [8192,512] f32
  int wave = (blockIdx.x * blockDim.x + threadIdx.x) >> 5;
  int lane = threadIdx.x & 31;
  int mt = wave >> 5;            // 512 M-tiles
  int nt = wave & 31;            // 32 N-tiles
  int m  = lane & 15;
  int kg = lane >> 4;
  v8f acc = {};
  const unsigned short* arow = A + (mt * 16 + m) * H_;
  #pragma unroll
  for (int kt = 0; kt < 8; ++kt) {
    ABf a, bmat;
    a.u[0] = *(const U4*)(arow + kt * 32 + kg * 8);
    a.u[1] = *(const U4*)(arow + kt * 32 + 16 + kg * 8);
    const unsigned short* bt = Bp + (kt * 32 + nt) * 512 + lane * 16;
    bmat.u[0] = ((const U4*)bt)[0];
    bmat.u[1] = ((const U4*)bt)[1];
    acc = __builtin_amdgcn_wmma_f32_16x16x32_bf16(false, a.v, false, bmat.v,
                                                  (short)0, acc, false, false);
  }
  int n = nt * 16 + (lane & 15);
  int mbase = mt * 16 + 8 * (lane >> 4);
  #pragma unroll
  for (int r = 0; r < 8; ++r)
    P[(mbase + r) * P4_ + n] = acc[r];
}

// ---------------------------------------------------------------------------
// 5) Per-node scalar head: ws = h . W_w[0:256], wd = h . W_w[256:512].
// ---------------------------------------------------------------------------
__global__ void wdot_kernel(const unsigned short* __restrict__ Hbf,
                            const float* __restrict__ Ww,            // [512]
                            float* __restrict__ ws, float* __restrict__ wd) {
  int wave = (blockIdx.x * blockDim.x + threadIdx.x) >> 5;
  int lane = threadIdx.x & 31;
  if (wave >= BN_) return;
  U4 hu = *(const U4*)(Hbf + wave * H_ + lane * 8);
  unsigned int uu[4] = {hu.x, hu.y, hu.z, hu.w};
  float hv[8];
  #pragma unroll
  for (int t = 0; t < 4; ++t) {
    hv[2 * t]     = bf2f((unsigned short)(uu[t] & 0xFFFFu));
    hv[2 * t + 1] = bf2f((unsigned short)(uu[t] >> 16));
  }
  float s0 = 0.0f, s1 = 0.0f;
  int k = lane * 8;
  #pragma unroll
  for (int t = 0; t < 8; ++t) {
    s0 += hv[t] * Ww[k + t];
    s1 += hv[t] * Ww[H_ + k + t];
  }
  s0 = wave_sum(s0);
  s1 = wave_sum(s1);
  if (lane == 0) { ws[wave] = s0; wd[wave] = s1; }
}

// ---------------------------------------------------------------------------
// 6) Per-edge: Gaussian similarity + gumbel logit. One wave per edge;
//    each lane covers 4 of the 128 head dims. All node data stays L2-hot.
// ---------------------------------------------------------------------------
__global__ void edge_kernel(const float* __restrict__ P,
                            const float* __restrict__ ws, const float* __restrict__ wd,
                            const float* __restrict__ b_mean, const float* __restrict__ b_var,
                            const float* __restrict__ b_w, const float* __restrict__ gu,
                            float* __restrict__ sim, float* __restrict__ logit) {
  int b    = blockIdx.x;
  int wave = threadIdx.x >> 5;
  int lane = threadIdx.x & 31;
  int nw   = blockDim.x >> 5;
  const float bw = b_w[0];
  float4 bm  = ((const float4*)b_mean)[lane];
  float4 bvv = ((const float4*)b_var)[lane];
  for (int e = wave; e < E_; e += nw) {
    // invert upper-tri enumeration: e = i*(127-i)/2 + (j-i-1)
    int i = (int)floorf((127.0f - sqrtf(16129.0f - 8.0f * (float)e)) * 0.5f);
    if (i < 0) i = 0; if (i > 62) i = 62;
    while (i < 62 && (i + 1) * (127 - (i + 1)) / 2 <= e) ++i;
    while (i > 0 && i * (127 - i) / 2 > e) --i;
    int j  = e - i * (127 - i) / 2 + i + 1;
    int ni = b * N_ + i, nj = b * N_ + j;
    const float4* Pi = (const float4*)(P + ni * P4_);
    const float4* Pj = (const float4*)(P + nj * P4_);
    float4 ms = Pi[lane];        // mean_top (src)
    float4 md = Pj[32 + lane];   // mean_bot (dst)
    float4 vs = Pi[64 + lane];   // var_top  (src)
    float4 vd = Pj[96 + lane];   // var_bot  (dst)
    float m0 = ms.x + md.x + bm.x, v0 = softplusf(vs.x + vd.x + bvv.x) + 1e-6f;
    float m1 = ms.y + md.y + bm.y, v1 = softplusf(vs.y + vd.y + bvv.y) + 1e-6f;
    float m2 = ms.z + md.z + bm.z, v2 = softplusf(vs.z + vd.z + bvv.z) + 1e-6f;
    float m3 = ms.w + md.w + bm.w, v3 = softplusf(vs.w + vd.w + bvv.w) + 1e-6f;
    float acc = m0 * m0 / (v0 + 1e-8f) + m1 * m1 / (v1 + 1e-8f)
              + m2 * m2 / (v2 + 1e-8f) + m3 * m3 / (v3 + 1e-8f);
    acc = wave_sum(acc);
    if (lane == 0) {
      int ge = b * E_ + e;
      float s    = __expf(-0.5f * (acc * (1.0f / OUT_)));
      float wgt  = ws[ni] + wd[nj] + bw;
      float sigw = 1.0f / (1.0f + __expf(-wgt));
      float g    = -__logf(-__logf(gu[ge]));
      sim[ge]   = s;
      logit[ge] = (sigw + g) * 2.0f;    // / TEMPERATURE (0.5)
    }
  }
}

// ---------------------------------------------------------------------------
// 7) Deterministic two-stage softmax reductions over 258048 logits.
// ---------------------------------------------------------------------------
__global__ void reduce_max_kernel(const float* __restrict__ v, float* __restrict__ part, int n) {
  __shared__ float sm[8];
  float m = -3.4e38f;
  for (int idx = blockIdx.x * blockDim.x + threadIdx.x; idx < n; idx += gridDim.x * blockDim.x)
    m = fmaxf(m, v[idx]);
  m = wave_max(m);
  int w = threadIdx.x >> 5, lane = threadIdx.x & 31;
  if (lane == 0) sm[w] = m;
  __syncthreads();
  if (threadIdx.x < 32) {
    m = (lane < 8) ? sm[lane] : -3.4e38f;
    m = wave_max(m);
    if (lane == 0) part[blockIdx.x] = m;
  }
}

__global__ void finish_max_kernel(const float* __restrict__ part, float* __restrict__ gmax) {
  float m = fmaxf(part[threadIdx.x], part[threadIdx.x + 32]);
  m = wave_max(m);
  if (threadIdx.x == 0) *gmax = m;
}

__global__ void reduce_sum_kernel(const float* __restrict__ v, const float* __restrict__ gmax,
                                  float* __restrict__ part, int n) {
  __shared__ float sm[8];
  float mx = *gmax;
  float s = 0.0f;
  for (int idx = blockIdx.x * blockDim.x + threadIdx.x; idx < n; idx += gridDim.x * blockDim.x)
    s += __expf(v[idx] - mx);
  s = wave_sum(s);
  int w = threadIdx.x >> 5, lane = threadIdx.x & 31;
  if (lane == 0) sm[w] = s;
  __syncthreads();
  if (threadIdx.x < 32) {
    s = (lane < 8) ? sm[lane] : 0.0f;
    s = wave_sum(s);
    if (lane == 0) part[blockIdx.x] = s;
  }
}

__global__ void finish_sum_kernel(const float* __restrict__ part, float* __restrict__ gsum) {
  float s = part[threadIdx.x] + part[threadIdx.x + 32];
  s = wave_sum(s);
  if (threadIdx.x == 0) *gsum = s;
}

// ---------------------------------------------------------------------------
// 8) Scatter into dense adjacency [B,64,64]; zero everywhere else.
// ---------------------------------------------------------------------------
__global__ void final_kernel(const float* __restrict__ sim, const float* __restrict__ logit,
                             const float* __restrict__ gmax, const float* __restrict__ gsum,
                             float* __restrict__ out) {
  int tid = blockIdx.x * blockDim.x + threadIdx.x;
  if (tid >= B_ * N_ * N_) return;
  int b = tid >> 12, rem = tid & 4095;
  int r = rem >> 6, c = rem & 63;
  float val = 0.0f;
  if (r < c) {
    int ge = b * E_ + r * (127 - r) / 2 + (c - r - 1);
    val = sim[ge] * __expf(logit[ge] - *gmax) / *gsum;
  }
  out[tid] = val;
}

// ---------------------------------------------------------------------------
extern "C" void kernel_launch(void* const* d_in, const int* in_sizes, int n_in,
                              void* d_out, int out_size, void* d_ws, size_t ws_size,
                              hipStream_t stream) {
  (void)in_sizes; (void)n_in; (void)out_size; (void)ws_size;
  const float* topo = (const float*)d_in[0];
  const float* temp = (const float*)d_in[1];
  const float* gu   = (const float*)d_in[2];
  const float* Wg   = (const float*)d_in[3];
  const float* bg   = (const float*)d_in[4];
  const float* Wm   = (const float*)d_in[5];
  const float* bm   = (const float*)d_in[6];
  const float* Wv   = (const float*)d_in[7];
  const float* bv   = (const float*)d_in[8];
  const float* Ww   = (const float*)d_in[9];
  const float* bw   = (const float*)d_in[10];
  float* out = (float*)d_out;

  char* p = (char*)d_ws;
  auto alloc = [&](size_t bytes) -> char* {
    char* r = p;
    p += (bytes + 255) & ~size_t(255);
    return r;
  };
  unsigned short* agg = (unsigned short*)alloc((size_t)BN_ * IN_ * 2);
  unsigned short* hbf = (unsigned short*)alloc((size_t)BN_ * H_ * 2);
  unsigned short* WgP = (unsigned short*)alloc((size_t)10 * 16 * 512 * 2);
  unsigned short* W4P = (unsigned short*)alloc((size_t)8 * 32 * 512 * 2);
  float* P    = (float*)alloc((size_t)BN_ * P4_ * 4);
  float* wsv  = (float*)alloc((size_t)BN_ * 4);
  float* wdv  = (float*)alloc((size_t)BN_ * 4);
  float* simA = (float*)alloc((size_t)BE_ * 4);
  float* logA = (float*)alloc((size_t)BE_ * 4);
  float* pmax = (float*)alloc(64 * 4);
  float* psum = (float*)alloc(64 * 4);
  float* gmax = (float*)alloc(4);
  float* gsum = (float*)alloc(4);

  prefix_mean_kernel<<<B_, IN_, 0, stream>>>(topo, temp, agg);
  pack_wgnn_kernel<<<(10 * 16 * 512 + 255) / 256, 256, 0, stream>>>(Wg, WgP);
  pack_w4_kernel<<<(8 * 32 * 512 + 255) / 256, 256, 0, stream>>>(Wm, Wv, W4P);
  gemm1_kernel<<<(512 * 16) / 4, 128, 0, stream>>>(agg, WgP, bg, hbf);
  gemm2_kernel<<<(512 * 32) / 4, 128, 0, stream>>>(hbf, W4P, P);
  wdot_kernel<<<BN_ / 8, 256, 0, stream>>>(hbf, Ww, wsv, wdv);
  edge_kernel<<<B_, 256, 0, stream>>>(P, wsv, wdv, bm, bv, bw, gu, simA, logA);
  reduce_max_kernel<<<64, 256, 0, stream>>>(logA, pmax, BE_);
  finish_max_kernel<<<1, 32, 0, stream>>>(pmax, gmax);
  reduce_sum_kernel<<<64, 256, 0, stream>>>(logA, gmax, psum, BE_);
  finish_sum_kernel<<<1, 32, 0, stream>>>(psum, gsum);
  final_kernel<<<(B_ * N_ * N_ + 255) / 256, 256, 0, stream>>>(simA, logA, gmax, gsum, out);
}
